// CNN_transformer_hr_xyz_41051297415299
// MI455X (gfx1250) — matrix operand order repaired and checked by reading, not compiled, not run on hardware
//
#include <hip/hip_runtime.h>
#include <math.h>

// ---------------- problem constants ----------------
#define BB   32
#define CC   512      // channels after conv == attention sequence length
#define DD   512      // conv spatial axis == attention feature dim
#define DPAD 520      // 6 leading zero cols (causal halo) + 512 + 2 tail pad
#define HH   8
#define DH   64
#define DFF  2048
#define KT   7
#define CD   (CC * DD)
#define BCD  (BB * CD)
#define BCDP (BB * CC * DPAD)

typedef __attribute__((ext_vector_type(16))) __bf16 v16bf;
typedef __attribute__((ext_vector_type(8)))  __bf16 v8bf;
typedef __attribute__((ext_vector_type(8)))  float  v8f;
typedef __attribute__((ext_vector_type(4)))  unsigned int v4u;
typedef __attribute__((ext_vector_type(8)))  int v8i;
typedef __attribute__((ext_vector_type(4)))  int v4i;

#if __has_builtin(__builtin_amdgcn_tensor_load_to_lds)
#define HAVE_TDM 1
#else
#define HAVE_TDM 0
#endif

// ---------------- WMMA operand helpers (wave32 layouts per CDNA5 ISA) -------
__device__ __forceinline__ v16bf bf_combine(v8bf a, v8bf b) {
    union { v16bf v; v8bf h[2]; } u;
    u.h[0] = a; u.h[1] = b;
    return u.v;
}

// A tile in LDS: [m][k], 64 rows x 32 k, k contiguous.
__device__ __forceinline__ v16bf lds_read_A(const __bf16* lA, int wave, int lane) {
    const int m   = wave * 16 + (lane & 15);
    const int off = (lane < 16) ? 0 : 8;
    const __bf16* p = lA + m * 32;
    return bf_combine(*(const v8bf*)(p + off), *(const v8bf*)(p + 16 + off));
}

// B tile in LDS: [n][k] rows; optional row shift (causal tap offset).
__device__ __forceinline__ v16bf lds_read_B(const __bf16* lB, int nt, int lane, int shift) {
    const int n   = nt * 16 + (lane & 15) + shift;
    const int off = (lane < 16) ? 0 : 8;
    const __bf16* p = lB + n * 32 + off;
    return bf_combine(*(const v8bf*)p, *(const v8bf*)(p + 16));
}

#if HAVE_TDM
// TDM: DMA a 64-row x 32-col bf16 tile (global row stride 512 elems) into LDS.
__device__ __forceinline__ void tdm_load_tile_a(const __bf16* gsrc, __bf16* lds_dst) {
    const unsigned lds = (unsigned)(unsigned long long)(void*)lds_dst;
    const unsigned long long ga = (unsigned long long)gsrc;
    v4u g0;
    g0[0] = 1u;                                            // count=1 (valid user D#)
    g0[1] = lds;                                           // lds_addr
    g0[2] = (unsigned)(ga & 0xFFFFFFFFu);                  // global_addr[31:0]
    g0[3] = (unsigned)((ga >> 32) & 0x1FFFFFFu) | (2u << 30); // addr[56:32] | type=2
    v8i g1;
    g1[0] = (int)(1u << 16);                               // data_size = 2 bytes
    g1[1] = (int)((512u & 0xFFFFu) << 16);                 // tensor_dim0 lo16 @ bits63:48
    g1[2] = (int)((512u & 0xFFFFu) << 16);                 // dim0 hi16=0 | tensor_dim1 lo16
    g1[3] = (int)(32u << 16);                              // dim1 hi16=0 | tile_dim0=32
    g1[4] = 64;                                            // tile_dim1=64
    g1[5] = 512;                                           // tensor_dim0_stride lo32
    g1[6] = 0;
    g1[7] = 0;
    v4i z4 = {0, 0, 0, 0};
#if defined(__clang_major__) && (__clang_major__ >= 23)
    v8i z8 = {0, 0, 0, 0, 0, 0, 0, 0};
    __builtin_amdgcn_tensor_load_to_lds(g0, g1, z4, z4, z8, 0);
#else
    __builtin_amdgcn_tensor_load_to_lds(g0, g1, z4, z4, 0);
#endif
}
#endif

// ---------------- elementwise kernels ----------------
__global__ void cast_bf16_kernel(const float* __restrict__ src,
                                 __bf16* __restrict__ dst, int n) {
    int i = blockIdx.x * 256 + threadIdx.x;
    if (i < n) dst[i] = (__bf16)src[i];
}

// f32 (B,C,D) -> padded bf16 (B,C,DPAD): col = d+6, halo/tail zeroed
__global__ void cast_pad_kernel(const float* __restrict__ src,
                                __bf16* __restrict__ dst) {
    int i = blockIdx.x * 256 + threadIdx.x;
    if (i < BCDP) {
        int col = i % DPAD;
        int row = i / DPAD;
        int d = col - 6;
        float v = (d >= 0 && d < DD) ? src[(size_t)row * DD + d] : 0.0f;
        dst[i] = (__bf16)v;
    }
}

// zero the 8 pad columns of a padded (B*C, DPAD) buffer
__global__ void pad_zero_kernel(__bf16* __restrict__ buf) {
    int i = blockIdx.x * 256 + threadIdx.x;
    if (i < BB * CC * 8) {
        int r = i >> 3, c = i & 7;
        int col = (c < 6) ? c : (DD + c);   // cols 0..5, 518, 519
        buf[(size_t)r * DPAD + col] = (__bf16)0.0f;
    }
}

// (CO,CI,K) f32 -> (K,CO,CI) bf16
__global__ void prep_convw_kernel(const float* __restrict__ src,
                                  __bf16* __restrict__ dst) {
    int i = blockIdx.x * 256 + threadIdx.x;
    if (i < CC * CC * KT) {
        int t  = i % KT;
        int ci = (i / KT) % CC;
        int co = i / (KT * CC);
        dst[(size_t)t * CC * CC + (size_t)co * CC + ci] = (__bf16)src[i];
    }
}

// rows of 512; torch-style g*(x-mean)/(sqrt(var_unbiased)+eps)+b -> bf16
__global__ __launch_bounds__(256)
void layernorm_bf16_kernel(const float* __restrict__ x,
                           const float* __restrict__ g,
                           const float* __restrict__ b,
                           __bf16* __restrict__ out, int padOut) {
    const int row  = blockIdx.x * 8 + (threadIdx.x >> 5);
    const int lane = threadIdx.x & 31;
    const float* xr = x + (size_t)row * DD;
    float v[16];
    float s = 0.0f;
#pragma unroll
    for (int i = 0; i < 16; ++i) { v[i] = xr[lane + i * 32]; s += v[i]; }
#pragma unroll
    for (int m = 16; m >= 1; m >>= 1) s += __shfl_xor(s, m);
    const float mean = s * (1.0f / (float)DD);
    float ss = 0.0f;
#pragma unroll
    for (int i = 0; i < 16; ++i) { float d = v[i] - mean; ss += d * d; }
#pragma unroll
    for (int m = 16; m >= 1; m >>= 1) ss += __shfl_xor(ss, m);
    const float var = ss * (1.0f / (float)(DD - 1));   // ddof=1
    const float inv = 1.0f / (sqrtf(var) + 1e-6f);
    __bf16* orow = out + (padOut ? ((size_t)row * DPAD + 6) : ((size_t)row * DD));
#pragma unroll
    for (int i = 0; i < 16; ++i) {
        int c = lane + i * 32;
        orow[c] = (__bf16)(g[c] * (v[i] - mean) * inv + b[c]);
    }
}

// softmax over rows of 512 (H*C rows of one batch), f32 -> bf16
__global__ __launch_bounds__(256)
void softmax_bf16_kernel(const float* __restrict__ S, __bf16* __restrict__ P) {
    const int row  = blockIdx.x * 8 + (threadIdx.x >> 5);
    const int lane = threadIdx.x & 31;
    const float* sr = S + (size_t)row * CC;
    float v[16];
    float mx = -3.0e38f;
#pragma unroll
    for (int i = 0; i < 16; ++i) { v[i] = sr[lane + i * 32]; mx = fmaxf(mx, v[i]); }
#pragma unroll
    for (int m = 16; m >= 1; m >>= 1) mx = fmaxf(mx, __shfl_xor(mx, m));
    float s = 0.0f;
#pragma unroll
    for (int i = 0; i < 16; ++i) { v[i] = __expf(v[i] - mx); s += v[i]; }
#pragma unroll
    for (int m = 16; m >= 1; m >>= 1) s += __shfl_xor(s, m);
    const float inv = 1.0f / s;
#pragma unroll
    for (int i = 0; i < 16; ++i)
        P[(size_t)row * CC + lane + i * 32] = (__bf16)(v[i] * inv);
}

// ---------------- causal conv as GEMM (padded input, branch-free) ----------
// y[b,co,d] = scale * ( bias[co] + sum_{tap,ci} Wt[tap][co][ci] * xpad[b,ci,d+tap] )
__global__ __launch_bounds__(128)
void conv_gemm_kernel(const __bf16* __restrict__ X,     // (B, CI, DPAD) bf16
                      const __bf16* __restrict__ Wt,    // (K, CO, CI) bf16
                      const float*  __restrict__ bias,  // (CO)
                      float*  __restrict__ outF,        // (B, CO, DD) or null
                      __bf16* __restrict__ outB,        // padded (B,CO,DPAD) or null
                      float scale, int padOut) {
    __shared__ __attribute__((aligned(16))) __bf16 lAa[64 * 32];
    __shared__ __attribute__((aligned(16))) __bf16 lAb[64 * 32];
    __shared__ __attribute__((aligned(16))) __bf16 lB[72 * 32];
    const int tid  = threadIdx.x;
    const int wave = tid >> 5, lane = tid & 31;
    const int n0 = blockIdx.x * 64;        // d tile
    const int m0 = blockIdx.y * 64;        // co tile
    const int b  = blockIdx.z;
    const __bf16* Xb = X + (size_t)b * CC * DPAD;

    v8f acc[4];
#pragma unroll
    for (int i = 0; i < 4; ++i) acc[i] = 0.0f;

    for (int kc = 0; kc < CC / 32; ++kc) {
#if HAVE_TDM
        if (wave == 0)   // prime the A pipeline: DMA tap 0 while we stage B
            tdm_load_tile_a(Wt + (size_t)m0 * CC + kc * 32, lAa);
#endif
        // Stage one 72-wide input stripe [n=d][k=ci]; all 7 taps reuse it.
        {
            const int k = tid & 31;
            const int g = tid >> 5;
            const __bf16* gp = Xb + (size_t)(kc * 32 + k) * DPAD + n0 + g * 16;
            __builtin_prefetch(gp + DPAD, 0, 0);
            v8bf x0 = *(const v8bf*)gp, x1 = *(const v8bf*)(gp + 8);
#pragma unroll
            for (int c = 0; c < 8; ++c) {
                lB[(g * 16 + c) * 32 + k]     = x0[c];
                lB[(g * 16 + 8 + c) * 32 + k] = x1[c];
            }
            if (g == 0) {   // halo columns 64..71
                v8bf x2 = *(const v8bf*)(gp + 64);
#pragma unroll
                for (int c = 0; c < 8; ++c) lB[(64 + c) * 32 + k] = x2[c];
            }
        }
        for (int tap = 0; tap < KT; ++tap) {
            __bf16* cur = (tap & 1) ? lAb : lAa;
#if HAVE_TDM
            if (wave == 0) {
                if (tap < KT - 1) {
                    __bf16* nxt = (tap & 1) ? lAa : lAb;
                    tdm_load_tile_a(Wt + (size_t)(tap + 1) * CC * CC +
                                        (size_t)m0 * CC + kc * 32, nxt);
                    __builtin_amdgcn_s_wait_tensorcnt(1);  // current tile done
                } else {
                    __builtin_amdgcn_s_wait_tensorcnt(0);  // drain pipeline
                }
            }
#else
            {   // fallback: vector-load the weight tile
                int row = tid >> 1, half = tid & 1;
                const __bf16* gp = Wt + (size_t)tap * CC * CC +
                                   (size_t)(m0 + row) * CC + kc * 32 + half * 16;
                *(v8bf*)(cur + row * 32 + half * 16)     = *(const v8bf*)gp;
                *(v8bf*)(cur + row * 32 + half * 16 + 8) = *(const v8bf*)(gp + 8);
            }
#endif
            __syncthreads();
            v16bf a = lds_read_A(cur, wave, lane);
#pragma unroll
            for (int nt = 0; nt < 4; ++nt)
                acc[nt] = __builtin_amdgcn_wmma_f32_16x16x32_bf16(
                    false, a, false, lds_read_B(lB, nt, lane, tap), (short)0,
                    acc[nt], false, false);
            __syncthreads();
        }
    }
#pragma unroll
    for (int nt = 0; nt < 4; ++nt) {
        const int col   = n0 + nt * 16 + (lane & 15);
        const int rbase = m0 + wave * 16 + ((lane < 16) ? 0 : 8);
#pragma unroll
        for (int r = 0; r < 8; ++r) {
            const int row = rbase + r;
            const float v = (acc[nt][r] + bias[row]) * scale;
            if (outF) outF[((size_t)b * CC + row) * DD + col] = v;
            if (outB) {
                const size_t idx = padOut ? (((size_t)b * CC + row) * DPAD + 6 + col)
                                          : (((size_t)b * CC + row) * DD + col);
                outB[idx] = (__bf16)v;
            }
        }
    }
}

// ---------------- attention scores: S = (Q Kh^T) / 8, per batch ------------
__global__ __launch_bounds__(128)
void attn_scores_kernel(const __bf16* __restrict__ Q,
                        const __bf16* __restrict__ Kmat,
                        float* __restrict__ S) {
    __shared__ __attribute__((aligned(16))) __bf16 lA[64 * 32];
    __shared__ __attribute__((aligned(16))) __bf16 lB[64 * 32];
    const int tid  = threadIdx.x;
    const int wave = tid >> 5, lane = tid & 31;
    const int j0 = blockIdx.x * 64;
    const int i0 = blockIdx.y * 64;
    const int h  = blockIdx.z;

    v8f acc[4];
#pragma unroll
    for (int i = 0; i < 4; ++i) acc[i] = 0.0f;

    for (int kc = 0; kc < DH / 32; ++kc) {
        {
            int row = tid >> 1, half = tid & 1;
            const __bf16* gp = Q + (size_t)(i0 + row) * DD + h * DH + kc * 32 + half * 16;
            *(v8bf*)(lA + row * 32 + half * 16)     = *(const v8bf*)gp;
            *(v8bf*)(lA + row * 32 + half * 16 + 8) = *(const v8bf*)(gp + 8);
        }
        {
            int n = tid >> 1, half = tid & 1;
            const __bf16* gp = Kmat + (size_t)(j0 + n) * DD + h * DH + kc * 32 + half * 16;
            *(v8bf*)(lB + n * 32 + half * 16)     = *(const v8bf*)gp;
            *(v8bf*)(lB + n * 32 + half * 16 + 8) = *(const v8bf*)(gp + 8);
        }
        __syncthreads();
        v16bf a = lds_read_A(lA, wave, lane);
#pragma unroll
        for (int nt = 0; nt < 4; ++nt)
            acc[nt] = __builtin_amdgcn_wmma_f32_16x16x32_bf16(
                false, a, false, lds_read_B(lB, nt, lane, 0), (short)0,
                acc[nt], false, false);
        __syncthreads();
    }
#pragma unroll
    for (int nt = 0; nt < 4; ++nt) {
        const int col   = j0 + nt * 16 + (lane & 15);
        const int rbase = i0 + wave * 16 + ((lane < 16) ? 0 : 8);
#pragma unroll
        for (int r = 0; r < 8; ++r)
            S[((size_t)h * CC + rbase + r) * CC + col] = acc[nt][r] * 0.125f;
    }
}

// ---------------- attention O = P @ V (padded O for the next conv) ---------
__global__ __launch_bounds__(128)
void attn_pv_kernel(const __bf16* __restrict__ P,   // (H, C, C)
                    const __bf16* __restrict__ V,   // (C, D) batch b
                    __bf16* __restrict__ O) {       // padded (C, DPAD) batch b
    __shared__ __attribute__((aligned(16))) __bf16 lA[64 * 32];
    __shared__ __attribute__((aligned(16))) __bf16 lB[64 * 32];
    const int tid  = threadIdx.x;
    const int wave = tid >> 5, lane = tid & 31;
    const int i0 = blockIdx.y * 64;
    const int h  = blockIdx.z;

    v8f acc[4];
#pragma unroll
    for (int i = 0; i < 4; ++i) acc[i] = 0.0f;

    for (int kc = 0; kc < CC / 32; ++kc) {
        {
            int row = tid >> 1, half = tid & 1;
            const __bf16* gp = P + ((size_t)h * CC + i0 + row) * CC + kc * 32 + half * 16;
            *(v8bf*)(lA + row * 32 + half * 16)     = *(const v8bf*)gp;
            *(v8bf*)(lA + row * 32 + half * 16 + 8) = *(const v8bf*)(gp + 8);
        }
        {
            int k = tid >> 2, nb = tid & 3;
            const __bf16* gp = V + (size_t)(kc * 32 + k) * DD + h * DH + nb * 16;
            v8bf x0 = *(const v8bf*)gp, x1 = *(const v8bf*)(gp + 8);
#pragma unroll
            for (int c = 0; c < 8; ++c) {
                lB[(nb * 16 + c) * 32 + k]     = x0[c];
                lB[(nb * 16 + 8 + c) * 32 + k] = x1[c];
            }
        }
        __syncthreads();
        v16bf a = lds_read_A(lA, wave, lane);
#pragma unroll
        for (int nt = 0; nt < 4; ++nt)
            acc[nt] = __builtin_amdgcn_wmma_f32_16x16x32_bf16(
                false, a, false, lds_read_B(lB, nt, lane, 0), (short)0,
                acc[nt], false, false);
        __syncthreads();
    }
#pragma unroll
    for (int nt = 0; nt < 4; ++nt) {
        const int col   = h * DH + nt * 16 + (lane & 15);
        const int rbase = i0 + wave * 16 + ((lane < 16) ? 0 : 8);
#pragma unroll
        for (int r = 0; r < 8; ++r)
            O[(size_t)(rbase + r) * DPAD + 6 + col] = (__bf16)acc[nt][r];
    }
}

// ---------------- FFN GEMM: out = scale * act(A @ Bm + bias[col]) ----------
__global__ __launch_bounds__(128)
void ffn_gemm_kernel(const __bf16* __restrict__ A,
                     const __bf16* __restrict__ Bm,
                     const float*  __restrict__ bias,
                     float*  __restrict__ outF,
                     __bf16* __restrict__ outB,
                     int Kdim, int Ndim, float scale, int doRelu) {
    __shared__ __attribute__((aligned(16))) __bf16 lA[64 * 32];
    __shared__ __attribute__((aligned(16))) __bf16 lB[64 * 32];
    const int tid  = threadIdx.x;
    const int wave = tid >> 5, lane = tid & 31;
    const int n0 = blockIdx.x * 64;
    const int m0 = blockIdx.y * 64;

    v8f acc[4];
#pragma unroll
    for (int i = 0; i < 4; ++i) acc[i] = 0.0f;

    for (int kc = 0; kc < Kdim / 32; ++kc) {
        {
            int row = tid >> 1, half = tid & 1;
            const __bf16* gp = A + (size_t)(m0 + row) * Kdim + kc * 32 + half * 16;
            __builtin_prefetch(gp + 32, 0, 0);
            *(v8bf*)(lA + row * 32 + half * 16)     = *(const v8bf*)gp;
            *(v8bf*)(lA + row * 32 + half * 16 + 8) = *(const v8bf*)(gp + 8);
        }
        {
            int k = tid >> 2, nb = tid & 3;
            const __bf16* gp = Bm + (size_t)(kc * 32 + k) * Ndim + n0 + nb * 16;
            v8bf x0 = *(const v8bf*)gp, x1 = *(const v8bf*)(gp + 8);
#pragma unroll
            for (int c = 0; c < 8; ++c) {
                lB[(nb * 16 + c) * 32 + k]     = x0[c];
                lB[(nb * 16 + 8 + c) * 32 + k] = x1[c];
            }
        }
        __syncthreads();
        v16bf a = lds_read_A(lA, wave, lane);
#pragma unroll
        for (int nt = 0; nt < 4; ++nt)
            acc[nt] = __builtin_amdgcn_wmma_f32_16x16x32_bf16(
                false, a, false, lds_read_B(lB, nt, lane, 0), (short)0,
                acc[nt], false, false);
        __syncthreads();
    }
#pragma unroll
    for (int nt = 0; nt < 4; ++nt) {
        const int col   = n0 + nt * 16 + (lane & 15);
        const int rbase = m0 + wave * 16 + ((lane < 16) ? 0 : 8);
#pragma unroll
        for (int r = 0; r < 8; ++r) {
            float v = acc[nt][r] + bias[col];
            if (doRelu) v = fmaxf(v, 0.0f);
            v *= scale;
            const size_t idx = (size_t)(rbase + r) * Ndim + col;
            if (outF) outF[idx] = v;
            if (outB) outB[idx] = (__bf16)v;
        }
    }
}

// ---------------- host orchestration ----------------
extern "C" void kernel_launch(void* const* d_in, const int* in_sizes, int n_in,
                              void* d_out, int out_size, void* d_ws, size_t ws_size,
                              hipStream_t stream) {
    const float* x         = (const float*)d_in[0];
    const float* w_conv_in = (const float*)d_in[1];
    const float* b_conv_in = (const float*)d_in[2];
    const float* wq        = (const float*)d_in[3];
    const float* bq        = (const float*)d_in[4];
    const float* wk        = (const float*)d_in[5];
    const float* bk        = (const float*)d_in[6];
    const float* wv        = (const float*)d_in[7];
    const float* bv        = (const float*)d_in[8];
    const float* wo        = (const float*)d_in[9];
    const float* bo        = (const float*)d_in[10];
    const float* ln0_g     = (const float*)d_in[11];
    const float* ln0_b     = (const float*)d_in[12];
    const float* ln1_g     = (const float*)d_in[13];
    const float* ln1_b     = (const float*)d_in[14];
    const float* w1        = (const float*)d_in[15];
    const float* b1        = (const float*)d_in[16];
    const float* w2        = (const float*)d_in[17];
    const float* b2        = (const float*)d_in[18];

    char* ws = (char*)d_ws;
    size_t off = 0;
    auto alloc = [&](size_t bytes) -> void* {
        void* p = ws + off;
        off += (bytes + 255) & ~(size_t)255;
        return p;
    };

    __bf16* xPad   = (__bf16*)alloc((size_t)BCDP * 2);
    __bf16* qinPad = (__bf16*)alloc((size_t)BCDP * 2);
    __bf16* xnPad  = (__bf16*)alloc((size_t)BCDP * 2);
    __bf16* oPad   = (__bf16*)alloc((size_t)BCDP * 2);
    __bf16* wt_in  = (__bf16*)alloc((size_t)KT * CC * CC * 2);
    __bf16* wt_q   = (__bf16*)alloc((size_t)KT * CC * CC * 2);
    __bf16* wt_k   = (__bf16*)alloc((size_t)KT * CC * CC * 2);
    __bf16* wt_v   = (__bf16*)alloc((size_t)KT * CC * CC * 2);
    __bf16* wt_o   = (__bf16*)alloc((size_t)KT * CC * CC * 2);
    __bf16* w1bf   = (__bf16*)alloc((size_t)DD * DFF * 2);
    __bf16* w2bf   = (__bf16*)alloc((size_t)DFF * DD * 2);
    float*  qinF   = (float*) alloc((size_t)BCD * 4);
    __bf16* qB     = (__bf16*)alloc((size_t)BCD * 2);
    __bf16* kB     = (__bf16*)alloc((size_t)BCD * 2);
    __bf16* vB     = (__bf16*)alloc((size_t)BCD * 2);
    float*  Sws    = (float*) alloc((size_t)HH * CC * CC * 4);
    __bf16* Pws    = (__bf16*)alloc((size_t)HH * CC * CC * 2);
    float*  h1F    = (float*) alloc((size_t)BCD * 4);
    __bf16* hnB    = (__bf16*)alloc((size_t)BCD * 2);
    __bf16* ff1B   = (__bf16*)alloc((size_t)BB * CC * DFF * 2);

    // 1) casts / pads / weight permutes
    cast_pad_kernel<<<(BCDP + 255) / 256, 256, 0, stream>>>(x, xPad);
    const int nPadz = BB * CC * 8;
    pad_zero_kernel<<<(nPadz + 255) / 256, 256, 0, stream>>>(qinPad);
    pad_zero_kernel<<<(nPadz + 255) / 256, 256, 0, stream>>>(xnPad);
    pad_zero_kernel<<<(nPadz + 255) / 256, 256, 0, stream>>>(oPad);
    const int nW = CC * CC * KT;
    prep_convw_kernel<<<(nW + 255) / 256, 256, 0, stream>>>(w_conv_in, wt_in);
    prep_convw_kernel<<<(nW + 255) / 256, 256, 0, stream>>>(wq, wt_q);
    prep_convw_kernel<<<(nW + 255) / 256, 256, 0, stream>>>(wk, wt_k);
    prep_convw_kernel<<<(nW + 255) / 256, 256, 0, stream>>>(wv, wt_v);
    prep_convw_kernel<<<(nW + 255) / 256, 256, 0, stream>>>(wo, wt_o);
    cast_bf16_kernel<<<(DD * DFF + 255) / 256, 256, 0, stream>>>(w1, w1bf, DD * DFF);
    cast_bf16_kernel<<<(DFF * DD + 255) / 256, 256, 0, stream>>>(w2, w2bf, DFF * DD);

    dim3 convGrid(DD / 64, CC / 64, BB);
    // 2) query_in = conv(x)
    conv_gemm_kernel<<<convGrid, 128, 0, stream>>>(xPad, wt_in, b_conv_in,
                                                   qinF, qinPad, 1.0f, 1);
    // 3) xn = LN0(query_in)
    layernorm_bf16_kernel<<<(BB * CC) / 8, 256, 0, stream>>>(qinF, ln0_g, ln0_b, xnPad, 1);
    // 4) q = conv(query_in), k = conv(xn), v = conv(xn)
    conv_gemm_kernel<<<convGrid, 128, 0, stream>>>(qinPad, wt_q, bq, nullptr, qB, 1.0f, 0);
    conv_gemm_kernel<<<convGrid, 128, 0, stream>>>(xnPad,  wt_k, bk, nullptr, kB, 1.0f, 0);
    conv_gemm_kernel<<<convGrid, 128, 0, stream>>>(xnPad,  wt_v, bv, nullptr, vB, 1.0f, 0);
    // 5) attention, per batch (reuse S/P scratch)
    for (int b = 0; b < BB; ++b) {
        attn_scores_kernel<<<dim3(CC / 64, CC / 64, HH), 128, 0, stream>>>(
            qB + (size_t)b * CD, kB + (size_t)b * CD, Sws);
        softmax_bf16_kernel<<<(HH * CC) / 8, 256, 0, stream>>>(Sws, Pws);
        attn_pv_kernel<<<dim3(1, CC / 64, HH), 128, 0, stream>>>(
            Pws, vB + (size_t)b * CD, oPad + (size_t)b * CC * DPAD);
    }
    // 6) h1 = 2 * conv(o)
    conv_gemm_kernel<<<convGrid, 128, 0, stream>>>(oPad, wt_o, bo, h1F, nullptr, 2.0f, 0);
    // 7) hn = LN1(h1)
    layernorm_bf16_kernel<<<(BB * CC) / 8, 256, 0, stream>>>(h1F, ln1_g, ln1_b, hnB, 0);
    // 8) ff1 = relu(hn @ w1 + b1)
    ffn_gemm_kernel<<<dim3(DFF / 64, (BB * CC) / 64), 128, 0, stream>>>(
        hnB, w1bf, b1, nullptr, ff1B, DD, DFF, 1.0f, 1);
    // 9) out = 2 * (ff1 @ w2 + b2)
    ffn_gemm_kernel<<<dim3(DD / 64, (BB * CC) / 64), 128, 0, stream>>>(
        ff1B, w2bf, b2, (float*)d_out, nullptr, DFF, DD, 2.0f, 0);
}
